// Generator_56633438765214
// MI455X (gfx1250) — compile-verified, matrix operand order checked by reading, not run on hardware
//
#include <hip/hip_runtime.h>
#include <hip/hip_bf16.h>
#include <math.h>

// ---- problem constants (from reference) ----
constexpr int NS = 30, B = 16, T = 8, S = 400, H = 400, V = 20000, G = 3;
constexpr int BS = NS * B;        // 480 fused rows
constexpr int H3 = 3 * H;         // 1200
constexpr int Hp = 416;           // H padded to multiple of 32 (13 K-steps of 32)
constexpr int Sp = 416;           // S padded to multiple of 32
constexpr long long GATE_BASE = (long long)NS * B * T * V;  // 76,800,000

typedef __attribute__((ext_vector_type(16))) _Float16 v16h;
typedef __attribute__((ext_vector_type(8)))  float    v8f;

// ---------------- WMMA fragment helpers (16-bit, 16x16x32) ----------------
// A (MxK) / B (KxN, loaded transposed-source): lane l holds row/col (l&15);
// K elements: {koff+0..7, 16+koff+0..7}, koff = (l>>4)*8.  (ISA 7.12.2)
__device__ inline v16h frag_from_row(const _Float16* __restrict__ rowPtr, int k0, int lane) {
  int kb = k0 + ((lane >> 4) << 3);
  v16h r;
#pragma unroll
  for (int i = 0; i < 8; ++i) r[i] = rowPtr[kb + i];
#pragma unroll
  for (int i = 0; i < 8; ++i) r[i + 8] = rowPtr[kb + 16 + i];
  return r;
}
// B in natural row-major [K x N]: lane fixed at column, stride ldb per K.
__device__ inline v16h frag_from_col(const _Float16* __restrict__ colPtr, int ldb, int k0, int lane) {
  int kb = k0 + ((lane >> 4) << 3);
  v16h r;
#pragma unroll
  for (int i = 0; i < 8; ++i) r[i] = colPtr[(size_t)(kb + i) * ldb];
#pragma unroll
  for (int i = 0; i < 8; ++i) r[i + 8] = colPtr[(size_t)(kb + 16 + i) * ldb];
  return r;
}
__device__ inline v8f wmma_f16(v16h a, v16h b, v8f c) {
  return __builtin_amdgcn_wmma_f32_16x16x32_f16(false, a, false, b, (short)0, c, false, false);
}

// ---------------- staging / conversion kernels ----------------
// f32 [R x Cs] -> f16 [R x Cp], zero pad columns
__global__ __launch_bounds__(256) void k_cvt(const float* __restrict__ src,
                                             _Float16* __restrict__ dst,
                                             int R, int Cs, int Cp) {
  long long idx = (long long)blockIdx.x * blockDim.x + threadIdx.x;
  if (idx >= (long long)R * Cp) return;
  int r = (int)(idx / Cp), c = (int)(idx % Cp);
  dst[idx] = (c < Cs) ? (_Float16)src[(size_t)r * Cs + c] : (_Float16)0.0f;
}

// encoded_outputs f32 [B,S,H] -> f16 [B,Sp,Hp], zero pad rows+cols
__global__ __launch_bounds__(256) void k_cvt_enc(const float* __restrict__ enc,
                                                 _Float16* __restrict__ dst) {
  long long idx = (long long)blockIdx.x * blockDim.x + threadIdx.x;
  if (idx >= (long long)B * Sp * Hp) return;
  int b = (int)(idx / ((long long)Sp * Hp));
  int rem = (int)(idx % ((long long)Sp * Hp));
  int s = rem / Hp, c = rem % Hp;
  float v = (s < S && c < H) ? enc[((size_t)b * S + s) * H + c] : 0.0f;
  dst[idx] = (_Float16)v;
}

// X[t][i][:] = (t==0) ? slot_emb[i/B] : emb[target[i%B][i/B][t-1]]   -> f16 [T,BS,Hp]
__global__ __launch_bounds__(256) void k_build_x(const float* __restrict__ slot_emb,
                                                 const float* __restrict__ emb,
                                                 const int* __restrict__ target,
                                                 _Float16* __restrict__ x16) {
  long long idx = (long long)blockIdx.x * blockDim.x + threadIdx.x;
  if (idx >= (long long)T * BS * Hp) return;
  int t = (int)(idx / ((long long)BS * Hp));
  int rem = (int)(idx % ((long long)BS * Hp));
  int i = rem / Hp, c = rem % Hp;
  float v = 0.0f;
  if (c < H) {
    int slot = i / B, b = i % B;
    if (t == 0) v = slot_emb[(size_t)slot * H + c];
    else {
      int tok = target[((size_t)b * NS + slot) * T + (t - 1)];
      v = emb[(size_t)tok * H + c];
    }
  }
  x16[idx] = (_Float16)v;
}

// hid <- tile(encoded_hidden); also f16 copy with zero pad
__global__ __launch_bounds__(256) void k_init_hid(const float* __restrict__ eh,
                                                  float* __restrict__ hid32,
                                                  _Float16* __restrict__ hid16) {
  int idx = blockIdx.x * blockDim.x + threadIdx.x;
  if (idx >= BS * Hp) return;
  int i = idx / Hp, c = idx % Hp;
  float v = (c < H) ? eh[(size_t)(i % B) * H + c] : 0.0f;
  if (c < H) hid32[(size_t)i * H + c] = v;
  hid16[idx] = (_Float16)v;
}

// ------- main WMMA GEMM:  C[M,N] = A[M,Kp] * Bsrc[N,Kp]^T (+bias) -------
// One wave per 64x64 register-blocked tile (4 m-subtiles x 4 n-subtiles,
// 16 v8f accumulators ~ 170 VGPRs live). __launch_bounds__(32,1) tells the
// backend the true block size so it budgets the full per-wave VGPR file
// (no scratch spills around the WMMAs).
__global__ __launch_bounds__(32, 1) void k_gemm_bt(const _Float16* __restrict__ A,
                                                   const _Float16* __restrict__ Bm,
                                                   float* __restrict__ C,
                                                   const float* __restrict__ bias,
                                                   int M, int N, int Kp, int ldc) {
  const int lane = threadIdx.x;
  const int m0 = blockIdx.y * 64;
  const int n0 = blockIdx.x * 64;

  v8f acc[4][4];
#pragma unroll
  for (int mi = 0; mi < 4; ++mi)
#pragma unroll
    for (int nj = 0; nj < 4; ++nj)
#pragma unroll
      for (int i = 0; i < 8; ++i) acc[mi][nj][i] = 0.0f;

  const _Float16* arow[4];
  const _Float16* brow[4];
#pragma unroll
  for (int mi = 0; mi < 4; ++mi) {
    int m = m0 + mi * 16 + (lane & 15);
    if (m >= M) m = M - 1;                 // clamp (store is masked)
    arow[mi] = A + (size_t)m * Kp;
  }
#pragma unroll
  for (int nj = 0; nj < 4; ++nj) {
    int n = n0 + nj * 16 + (lane & 15);
    if (n >= N) n = N - 1;                 // clamp (store is masked)
    brow[nj] = Bm + (size_t)n * Kp;
  }

  for (int k0 = 0; k0 < Kp; k0 += 32) {
    // prefetch next k-chunk of B one iteration ahead (global_prefetch_b8)
    if (k0 + 32 < Kp) {
#pragma unroll
      for (int nj = 0; nj < 4; ++nj) __builtin_prefetch(brow[nj] + k0 + 32, 0, 1);
    }
    v16h af[4];
#pragma unroll
    for (int mi = 0; mi < 4; ++mi) af[mi] = frag_from_row(arow[mi], k0, lane);
#pragma unroll
    for (int nj = 0; nj < 4; ++nj) {
      v16h bf = frag_from_row(brow[nj], k0, lane);
#pragma unroll
      for (int mi = 0; mi < 4; ++mi)
        acc[mi][nj] = wmma_f16(af[mi], bf, acc[mi][nj]);
    }
  }

#pragma unroll
  for (int mi = 0; mi < 4; ++mi) {
    const int mb = m0 + mi * 16 + ((lane >> 4) << 3);
#pragma unroll
    for (int nj = 0; nj < 4; ++nj) {
      int n = n0 + nj * 16 + (lane & 15);
      if (n < N) {
        float bv = bias ? bias[n] : 0.0f;
#pragma unroll
        for (int i = 0; i < 8; ++i)
          if (mb + i < M) C[(size_t)(mb + i) * ldc + n] = acc[mi][nj][i] + bv;
      }
    }
  }
}

// ---------------- attention scores: per batch b, [NS x S] = h_new[slot*B+b] . enc[b] ----------------
__global__ __launch_bounds__(32, 1) void k_scores(const _Float16* __restrict__ hid16,
                                                  const _Float16* __restrict__ enc16,
                                                  float* __restrict__ scores) {
  const int lane = threadIdx.x;
  const int nt = blockIdx.x;   // 0..S/16-1
  const int mt = blockIdx.y;   // 0..1 (32 slot rows, mask to 30)
  const int bb = blockIdx.z;   // 0..B-1
  int slotA = mt * 16 + (lane & 15); if (slotA >= NS) slotA = NS - 1;
  const _Float16* arow = hid16 + (size_t)(slotA * B + bb) * Hp;
  const _Float16* brow = enc16 + ((size_t)bb * Sp + nt * 16 + (lane & 15)) * Hp;
  v8f acc;
#pragma unroll
  for (int i = 0; i < 8; ++i) acc[i] = 0.0f;
  for (int k0 = 0; k0 < Hp; k0 += 32)
    acc = wmma_f16(frag_from_row(arow, k0, lane), frag_from_row(brow, k0, lane), acc);
  const int n = nt * 16 + (lane & 15);
  const int mbase = mt * 16 + ((lane >> 4) << 3);
#pragma unroll
  for (int i = 0; i < 8; ++i) {
    int slot = mbase + i;
    if (slot < NS) scores[(size_t)(slot * B + bb) * S + n] = acc[i];
  }
}

// masked softmax over S, emit prob f32 and zero-padded f16
__global__ __launch_bounds__(128) void k_attn_softmax(const float* __restrict__ scores,
                                                      const int* __restrict__ lens,
                                                      float* __restrict__ prob32,
                                                      _Float16* __restrict__ prob16) {
  const int i = blockIdx.x;
  const int len = lens[i % B];
  __shared__ float red[128];
  const float* srow = scores + (size_t)i * S;
  float mx = -INFINITY;
  for (int s = threadIdx.x; s < S; s += 128) if (s < len) mx = fmaxf(mx, srow[s]);
  red[threadIdx.x] = mx; __syncthreads();
  for (int off = 64; off > 0; off >>= 1) {
    if (threadIdx.x < off) red[threadIdx.x] = fmaxf(red[threadIdx.x], red[threadIdx.x + off]);
    __syncthreads();
  }
  mx = red[0]; __syncthreads();
  float sum = 0.0f;
  for (int s = threadIdx.x; s < S; s += 128) if (s < len) sum += expf(srow[s] - mx);
  red[threadIdx.x] = sum; __syncthreads();
  for (int off = 64; off > 0; off >>= 1) {
    if (threadIdx.x < off) red[threadIdx.x] += red[threadIdx.x + off];
    __syncthreads();
  }
  const float inv = 1.0f / red[0];
  for (int s = threadIdx.x; s < Sp; s += 128) {
    float p = (s < len) ? expf(srow[s] - mx) * inv : 0.0f;
    if (s < S) prob32[(size_t)i * S + s] = p;
    prob16[(size_t)i * Sp + s] = (_Float16)p;
  }
}

// context[slot*B+b][:] = prob . enc[b]  (B natural row-major, strided column loads)
__global__ __launch_bounds__(32, 1) void k_context(const _Float16* __restrict__ prob16,
                                                   const _Float16* __restrict__ enc16,
                                                   float* __restrict__ ctx) {
  const int lane = threadIdx.x;
  const int nt = blockIdx.x;   // 0..H/16-1
  const int mt = blockIdx.y;   // 0..1
  const int bb = blockIdx.z;
  int slotA = mt * 16 + (lane & 15); if (slotA >= NS) slotA = NS - 1;
  const _Float16* arow = prob16 + (size_t)(slotA * B + bb) * Sp;
  const _Float16* bcol = enc16 + (size_t)bb * Sp * Hp + nt * 16 + (lane & 15);
  v8f acc;
#pragma unroll
  for (int i = 0; i < 8; ++i) acc[i] = 0.0f;
  for (int k0 = 0; k0 < Sp; k0 += 32)
    acc = wmma_f16(frag_from_row(arow, k0, lane), frag_from_col(bcol, Hp, k0, lane), acc);
  const int n = nt * 16 + (lane & 15);
  const int mbase = mt * 16 + ((lane >> 4) << 3);
#pragma unroll
  for (int i = 0; i < 8; ++i) {
    int slot = mbase + i;
    if (slot < NS) ctx[(size_t)(slot * B + bb) * H + n] = acc[i];
  }
}

// GRU elementwise combine: h = (1-z)*n + z*h_prev  (in place, plus f16 copy)
__global__ __launch_bounds__(256) void k_gru(const float* __restrict__ gi_all,
                                             const float* __restrict__ gh,
                                             float* __restrict__ hid32,
                                             _Float16* __restrict__ hid16, int t) {
  int idx = blockIdx.x * blockDim.x + threadIdx.x;
  if (idx >= BS * H) return;
  int i = idx / H, j = idx % H;
  const float* gir = gi_all + ((size_t)t * BS + i) * H3;
  const float* ghr = gh + (size_t)i * H3;
  float r = 1.0f / (1.0f + expf(-(gir[j] + ghr[j])));
  float z = 1.0f / (1.0f + expf(-(gir[H + j] + ghr[H + j])));
  float n = tanhf(gir[2 * H + j] + r * ghr[2 * H + j]);
  float hn = (1.0f - z) * n + z * hid32[idx];
  hid32[idx] = hn;
  hid16[(size_t)i * Hp + j] = (_Float16)hn;
}

// switch = sigmoid(W_ratio . [h, ctx, x] + b), one wave per row
__global__ __launch_bounds__(32) void k_switch(const float* __restrict__ hid32,
                                               const float* __restrict__ ctx,
                                               const _Float16* __restrict__ x16,
                                               const float* __restrict__ Wr,
                                               const float* __restrict__ Wrb,
                                               float* __restrict__ sw, int t) {
  const int i = blockIdx.x, lane = threadIdx.x;
  const _Float16* xr = x16 + ((size_t)t * BS + i) * Hp;
  float acc = 0.0f;
  for (int j = lane; j < H3; j += 32) {
    float v;
    if (j < H) v = hid32[(size_t)i * H + j];
    else if (j < 2 * H) v = ctx[(size_t)i * H + (j - H)];
    else v = (float)xr[j - 2 * H];
    acc += Wr[j] * v;
  }
  for (int off = 16; off > 0; off >>= 1) acc += __shfl_down(acc, off);
  if (lane == 0) sw[i] = 1.0f / (1.0f + expf(-(acc + Wrb[0])));
}

// vocab softmax scaled by switch, write out[(i*T+t)*V + v]  (overwrite: keeps replay deterministic)
__global__ __launch_bounds__(256) void k_vocab_final(const float* __restrict__ logits,
                                                     const float* __restrict__ sw,
                                                     float* __restrict__ out, int t) {
  const int i = blockIdx.x;
  __shared__ float red[256];
  const float* lrow = logits + (size_t)i * V;
  float mx = -INFINITY;
  for (int v = threadIdx.x; v < V; v += 256) mx = fmaxf(mx, lrow[v]);
  red[threadIdx.x] = mx; __syncthreads();
  for (int off = 128; off > 0; off >>= 1) {
    if (threadIdx.x < off) red[threadIdx.x] = fmaxf(red[threadIdx.x], red[threadIdx.x + off]);
    __syncthreads();
  }
  mx = red[0]; __syncthreads();
  float sum = 0.0f;
  for (int v = threadIdx.x; v < V; v += 256) sum += expf(lrow[v] - mx);
  red[threadIdx.x] = sum; __syncthreads();
  for (int off = 128; off > 0; off >>= 1) {
    if (threadIdx.x < off) red[threadIdx.x] += red[threadIdx.x + off];
    __syncthreads();
  }
  const float scale = sw[i] / red[0];
  float* orow = out + ((size_t)i * T + t) * V;
  for (int v = threadIdx.x; v < V; v += 256) orow[v] = expf(lrow[v] - mx) * scale;
}

// pointer scatter: out[i,t,story[b][s]] += (1-switch)*prob  (after overwrite above)
__global__ __launch_bounds__(256) void k_scatter(const float* __restrict__ prob32,
                                                 const float* __restrict__ sw,
                                                 const int* __restrict__ story,
                                                 float* __restrict__ out, int t) {
  int idx = blockIdx.x * blockDim.x + threadIdx.x;
  if (idx >= BS * S) return;
  int i = idx / S, s = idx % S;
  float p = prob32[idx];
  int vi = story[(size_t)(i % B) * S + s];
  atomicAdd(out + ((size_t)i * T + t) * V + vi, (1.0f - sw[i]) * p);
}

// gate head (t==0 only): out[GATE_BASE + i*G + g] = ctx[i].W_gate[g] + b
__global__ __launch_bounds__(32) void k_gate(const float* __restrict__ ctx,
                                             const float* __restrict__ Wg,
                                             const float* __restrict__ bg,
                                             float* __restrict__ out) {
  const int i = blockIdx.x, lane = threadIdx.x;
  for (int g = 0; g < G; ++g) {
    float acc = 0.0f;
    for (int j = lane; j < H; j += 32) acc += ctx[(size_t)i * H + j] * Wg[(size_t)g * H + j];
    for (int off = 16; off > 0; off >>= 1) acc += __shfl_down(acc, off);
    if (lane == 0) out[GATE_BASE + (size_t)i * G + g] = acc + bg[g];
  }
}

// ---------------- host orchestration ----------------
extern "C" void kernel_launch(void* const* d_in, const int* in_sizes, int n_in,
                              void* d_out, int out_size, void* d_ws, size_t ws_size,
                              hipStream_t stream) {
  const float* encoded_hidden  = (const float*)d_in[0];
  const float* encoded_outputs = (const float*)d_in[1];
  const float* slot_emb        = (const float*)d_in[2];
  const float* emb             = (const float*)d_in[3];
  const float* W_ih            = (const float*)d_in[4];
  const float* W_hh            = (const float*)d_in[5];
  const float* b_ih            = (const float*)d_in[6];
  const float* b_hh            = (const float*)d_in[7];
  const float* W_ratio_w       = (const float*)d_in[8];
  const float* W_ratio_b       = (const float*)d_in[9];
  const float* W_gate_w        = (const float*)d_in[10];
  const float* W_gate_b        = (const float*)d_in[11];
  const int*   story           = (const int*)d_in[12];
  const int*   target          = (const int*)d_in[13];
  const int*   lens            = (const int*)d_in[14];
  float* out = (float*)d_out;

  char* w = (char*)d_ws;
  size_t off = 0;
  auto alloc = [&](size_t bytes) -> void* {
    void* p = w + off;
    off = (off + bytes + 255) & ~(size_t)255;
    return p;
  };
  _Float16* emb16  = (_Float16*)alloc((size_t)V * Hp * 2);
  _Float16* wih16  = (_Float16*)alloc((size_t)H3 * Hp * 2);
  _Float16* whh16  = (_Float16*)alloc((size_t)H3 * Hp * 2);
  _Float16* enc16  = (_Float16*)alloc((size_t)B * Sp * Hp * 2);
  _Float16* x16    = (_Float16*)alloc((size_t)T * BS * Hp * 2);
  float*    gi     = (float*)alloc((size_t)T * BS * H3 * 4);
  float*    hid32  = (float*)alloc((size_t)BS * H * 4);
  _Float16* hid16  = (_Float16*)alloc((size_t)BS * Hp * 2);
  float*    gh     = (float*)alloc((size_t)BS * H3 * 4);
  float*    scores = (float*)alloc((size_t)BS * S * 4);
  float*    prob32 = (float*)alloc((size_t)BS * S * 4);
  _Float16* prob16 = (_Float16*)alloc((size_t)BS * Sp * 2);
  float*    ctx    = (float*)alloc((size_t)BS * H * 4);
  float*    sw     = (float*)alloc((size_t)BS * 4);
  float*    logits = (float*)alloc((size_t)BS * V * 4);

  auto cdiv = [](long long a, long long b) { return (int)((a + b - 1) / b); };

  // Stage f16 operands (zero-padded K) and teacher-forced inputs.
  k_cvt<<<cdiv((long long)V * Hp, 256), 256, 0, stream>>>(emb, emb16, V, H, Hp);
  k_cvt<<<cdiv((long long)H3 * Hp, 256), 256, 0, stream>>>(W_ih, wih16, H3, H, Hp);
  k_cvt<<<cdiv((long long)H3 * Hp, 256), 256, 0, stream>>>(W_hh, whh16, H3, H, Hp);
  k_cvt_enc<<<cdiv((long long)B * Sp * Hp, 256), 256, 0, stream>>>(encoded_outputs, enc16);
  k_build_x<<<cdiv((long long)T * BS * Hp, 256), 256, 0, stream>>>(slot_emb, emb, target, x16);

  // Precompute gi for ALL steps in one WMMA GEMM: [T*BS, H3]
  k_gemm_bt<<<dim3(cdiv(H3, 64), cdiv(T * BS, 64)), 32, 0, stream>>>(
      x16, wih16, gi, b_ih, T * BS, H3, Hp, H3);

  k_init_hid<<<cdiv(BS * Hp, 256), 256, 0, stream>>>(encoded_hidden, hid32, hid16);

  for (int t = 0; t < T; ++t) {
    // gh = hid @ W_hh^T + b_hh
    k_gemm_bt<<<dim3(cdiv(H3, 64), cdiv(BS, 64)), 32, 0, stream>>>(
        hid16, whh16, gh, b_hh, BS, H3, Hp, H3);
    k_gru<<<cdiv(BS * H, 256), 256, 0, stream>>>(gi, gh, hid32, hid16, t);

    // attention
    k_scores<<<dim3(S / 16, 2, B), 32, 0, stream>>>(hid16, enc16, scores);
    k_attn_softmax<<<BS, 128, 0, stream>>>(scores, lens, prob32, prob16);
    k_context<<<dim3(H / 16, 2, B), 32, 0, stream>>>(prob16, enc16, ctx);
    k_switch<<<BS, 32, 0, stream>>>(hid32, ctx, x16, W_ratio_w, W_ratio_b, sw, t);

    // dominant GEMM: logits = h_new @ emb^T  (480 x 20000 x 416, f16 WMMA, 64x64 tiles)
    k_gemm_bt<<<dim3(cdiv(V, 64), cdiv(BS, 64)), 32, 0, stream>>>(
        hid16, emb16, logits, nullptr, BS, V, Hp, V);
    k_vocab_final<<<BS, 256, 0, stream>>>(logits, sw, out, t);
    k_scatter<<<cdiv(BS * S, 256), 256, 0, stream>>>(prob32, sw, story, out, t);

    if (t == 0) k_gate<<<BS, 32, 0, stream>>>(ctx, W_gate_w, W_gate_b, out);
  }
}
